// MHAPool2D_22436909154660
// MI455X (gfx1250) — compile-verified, matrix operand order checked by reading, not run on hardware
//
#include <hip/hip_runtime.h>
#include <cstdint>
#include <cstddef>

// ---------------- CDNA5 WMMA types ----------------
typedef __attribute__((ext_vector_type(16))) __bf16 v16bf;
typedef __attribute__((ext_vector_type(2)))  __bf16 v2bf;
typedef __attribute__((ext_vector_type(8)))  float  v8f;
typedef int v4i __attribute__((vector_size(16)));   // matches async-builtin param

union BF16x16 {            // 16 bf16 = one WMMA A/B operand (8 VGPRs)
    uint4          q[2];
    unsigned int   w[8];
    unsigned short s[16];
    v16bf          v;
};

// ---- bf16 conversion via native fptrunc (hardware cvt on gfx1250) ----
__device__ __forceinline__ unsigned short f2bf(float a) {
    __bf16 h = (__bf16)a;
    unsigned short r;
    __builtin_memcpy(&r, &h, 2);
    return r;
}

__device__ __forceinline__ unsigned int pack_bf(float a, float b) {
    v2bf p;
    p.x = (__bf16)a;
    p.y = (__bf16)b;
    unsigned int r;
    __builtin_memcpy(&r, &p, 4);
    return r;
}

__device__ __forceinline__ v8f wmma_bf16(v16bf a, v16bf b, v8f c) {
    return __builtin_amdgcn_wmma_f32_16x16x32_bf16(
        /*neg_a=*/false, a, /*neg_b=*/false, b,
        /*c_mod=*/(short)0, c, /*reuse_a=*/false, /*reuse_b=*/false);
}

// Load a 16-row x 32-(k)col bf16 tile (row stride STRIDE bf16) into the
// ISA A/B per-lane layout (CDNA5 ISA 7.12.2):
//   lane<16 : row = lane,     k = {0..7} then {16..23}
//   lane>=16: row = lane-16,  k = {8..15} then {24..31}
template <int STRIDE>
__device__ __forceinline__ v16bf tileA(const unsigned short* base, int lane) {
    const int row = lane & 15;
    const int lo  = (lane & 16) ? 8 : 0;
    BF16x16 u;
    u.q[0] = *(const uint4*)(base + row * STRIDE + lo);
    u.q[1] = *(const uint4*)(base + row * STRIDE + lo + 16);
    return u.v;
}

// ---------------- async global->LDS staging (gfx1250 path) ----------------
#if defined(__gfx1250__) && __has_builtin(__builtin_amdgcn_global_load_async_to_lds_b128)
#define USE_ASYNC_LDS 1
#endif

__device__ __forceinline__ void copy16_g2l(unsigned short* dst, const unsigned short* src) {
#ifdef USE_ASYNC_LDS
    __builtin_amdgcn_global_load_async_to_lds_b128(
        (v4i*)(void*)src, (v4i*)(void*)dst, /*offset=*/0, /*cpol=*/0);
#else
    *(uint4*)dst = *(const uint4*)src;
#endif
}

template <int N>
__device__ __forceinline__ void wait_stage() {
#ifdef USE_ASYNC_LDS
#if __has_builtin(__builtin_amdgcn_s_wait_asynccnt)
    __builtin_amdgcn_s_wait_asynccnt(N);
#else
    asm volatile("s_wait_asynccnt %0" ::"i"(N) : "memory");
#endif
#endif
}

// ---------------- problem constants ----------------
#define PB   4
#define PH   64
#define PW   64
#define PD   256
#define PN   8
#define PK   32
#define PV_  32
#define HW   (PH * PW)          // 4096
#define QHW  1024               // 32*32 pooled
#define NCOL (PN * PK)          // 256

// workspace layout (ushort units)
#define OFF_WKT  0
#define OFF_WVT  65536
#define OFF_WQT  131072
#define OFF_BLOB 196608                          // [16384][256] bf16
#define OFF_K    (OFF_BLOB + PB * HW * PD)       // [bn][4096][32] bf16
#define OFF_V    (OFF_K + PB * PN * HW * PK)     // [bn][32][4096] bf16 (transposed)
#define OFF_Q    (OFF_V + PB * PN * HW * PV_)    // [bn][1024][32] bf16

// ============ kernel 0a: weights -> bf16, transposed wT[col][d] ===========
__global__ __launch_bounds__(256) void prep_kernel(
    const float* __restrict__ wq, const float* __restrict__ wk,
    const float* __restrict__ wv, unsigned short* __restrict__ wqT,
    unsigned short* __restrict__ wkT, unsigned short* __restrict__ wvT) {
    int idx = blockIdx.x * 256 + threadIdx.x;     // 0 .. 65535
    int c = idx >> 8, d = idx & 255;
    wqT[c * 256 + d] = f2bf(wq[d * 256 + c]);
    wkT[c * 256 + d] = f2bf(wk[d * 256 + c]);
    wvT[c * 256 + d] = f2bf(wv[d * 256 + c]);
}

// ============ kernel 0b: blob f32 -> bf16 (one bandwidth pass) ============
__global__ __launch_bounds__(256) void blobcvt_kernel(
    const float* __restrict__ blob, unsigned short* __restrict__ blob_bf) {
    const size_t i = ((size_t)blockIdx.x * 256 + threadIdx.x) * 4;
    float4 f = *(const float4*)(blob + i);
    uint2 o;
    o.x = pack_bf(f.x, f.y);
    o.y = pack_bf(f.z, f.w);
    *(uint2*)(blob_bf + i) = o;
}

__device__ __forceinline__ void packA(BF16x16& u, int e, float4 f) {
    u.w[e >> 1]       = pack_bf(f.x, f.y);
    u.w[(e >> 1) + 1] = pack_bf(f.z, f.w);
}

__device__ __forceinline__ float4 f4max(float4 a, float4 b) {
    return make_float4(fmaxf(a.x, b.x), fmaxf(a.y, b.y),
                       fmaxf(a.z, b.z), fmaxf(a.w, b.w));
}

// ============ kernel 1: fused K+V projection (one pass over blob_bf) ======
// Each wave: 16 rows x 32 cols for BOTH wk and wv (4 accumulators, 4 WMMA/chunk)
__global__ __launch_bounds__(256) void proj_kv_kernel(
    const unsigned short* __restrict__ blob_bf,
    const unsigned short* __restrict__ wkT, const float* __restrict__ bk,
    const unsigned short* __restrict__ wvT, const float* __restrict__ bv,
    unsigned short* __restrict__ Kbf, unsigned short* __restrict__ Vt) {
    const int tid = threadIdx.x, lane = tid & 31, wave = tid >> 5;
    const int rb   = blockIdx.x * 16;                    // row base in [0,16384)
    const int col0 = wave * 32;
    const unsigned short* abase = blob_bf + (size_t)rb * PD;

    v8f aK0 = {0.f, 0.f, 0.f, 0.f, 0.f, 0.f, 0.f, 0.f};
    v8f aK1 = aK0, aV0 = aK0, aV1 = aK0;
#pragma unroll
    for (int d = 0; d < PD; d += 32) {
        const v16bf a = tileA<256>(abase + d, lane);     // pure bf16 loads, no cvt
        aK0 = wmma_bf16(a, tileA<256>(wkT + (size_t)col0 * 256 + d, lane), aK0);
        aK1 = wmma_bf16(a, tileA<256>(wkT + (size_t)(col0 + 16) * 256 + d, lane), aK1);
        aV0 = wmma_bf16(a, tileA<256>(wvT + (size_t)col0 * 256 + d, lane), aV0);
        aV1 = wmma_bf16(a, tileA<256>(wvT + (size_t)(col0 + 16) * 256 + d, lane), aV1);
    }
    const int cb = lane & 15, ro = (lane & 16) ? 8 : 0;
#pragma unroll
    for (int r = 0; r < 8; ++r) {
        const int rg = rb + r + ro;                      // global row
        const int bI = rg >> 12, pos = rg & 4095;
#pragma unroll
        for (int t = 0; t < 2; ++t) {
            const int cc = col0 + cb + 16 * t;
            const int nh = cc >> 5, kk = cc & 31;
            // K path
            float kval = (t ? aK1[r] : aK0[r]) + bk[cc];
            Kbf[(((size_t)bI * PN + nh) * HW + pos) * PK + kk] = f2bf(kval);
            // V path: swish, store transposed
            float vval = (t ? aV1[r] : aV0[r]) + bv[cc];
            vval = vval / (1.f + __expf(-vval));
            Vt[(((size_t)bI * PN + nh) * PV_ + kk) * HW + pos] = f2bf(vval);
        }
    }
}

// ============ kernel 2: pooled-Q projection ================================
__global__ __launch_bounds__(256) void qproj_kernel(
    const float* __restrict__ blob, const unsigned short* __restrict__ wqT,
    const float* __restrict__ bq, unsigned short* __restrict__ Qbf) {
    const int tid = threadIdx.x, lane = tid & 31, wave = tid >> 5;
    const int rb = blockIdx.x * 16;                      // pooled row base [0,4096)
    const int pr = rb + (lane & 15);
    const int bI0 = pr >> 10, p0 = pr & 1023, ph = p0 >> 5, pw = p0 & 31;
    const int lo = (lane & 16) ? 8 : 0;
    const int col0 = wave * 32;
    const float* r00 = blob + ((((size_t)bI0 * PH + 2 * ph) * PW) + 2 * pw) * PD;
    const float* r01 = r00 + PD;
    const float* r10 = r00 + (size_t)PW * PD;
    const float* r11 = r10 + PD;

    v8f acc0 = {0.f, 0.f, 0.f, 0.f, 0.f, 0.f, 0.f, 0.f};
    v8f acc1 = acc0;
#pragma unroll
    for (int d = 0; d < PD; d += 32) {
        BF16x16 ua;
        const int offs[4] = {lo, lo + 4, lo + 16, lo + 20};
#pragma unroll
        for (int g = 0; g < 4; ++g) {
            const int o = d + offs[g];
            float4 mx = f4max(f4max(*(const float4*)(r00 + o), *(const float4*)(r01 + o)),
                              f4max(*(const float4*)(r10 + o), *(const float4*)(r11 + o)));
            packA(ua, g * 4, mx);
        }
        acc0 = wmma_bf16(ua.v, tileA<256>(wqT + (size_t)col0 * 256 + d, lane), acc0);
        acc1 = wmma_bf16(ua.v, tileA<256>(wqT + (size_t)(col0 + 16) * 256 + d, lane), acc1);
    }
    const int cb = lane & 15, ro = (lane & 16) ? 8 : 0;
    const float scale = 0.17677669529663687f;            // 1/sqrt(32)
#pragma unroll
    for (int r = 0; r < 8; ++r) {
        const int rg = rb + r + ro;
        const int bI = rg >> 10, pp = rg & 1023;
#pragma unroll
        for (int t = 0; t < 2; ++t) {
            const int cc = col0 + cb + 16 * t;
            float val = ((t ? acc1[r] : acc0[r]) + bq[cc]) * scale;
            Qbf[(((size_t)bI * PN + (cc >> 5)) * QHW + pp) * PK + (cc & 31)] = f2bf(val);
        }
    }
}

// ============ kernel 3: fused flash cross-attention ========================
// block = 8 waves; each wave owns a 16-query tile; all waves stream the same
// 64-key K/V chunk staged in LDS (double buffered, async on gfx1250).
// Per chunk: 4 score WMMA + 4 PV WMMA; online softmax kept in registers.
__global__ __launch_bounds__(256) void attn_kernel(
    const unsigned short* __restrict__ Qbf, const unsigned short* __restrict__ Kbf,
    const unsigned short* __restrict__ Vt, float* __restrict__ out) {
    __shared__ unsigned short ldsK[2][2048];   // 64 keys x 32 kdim bf16
    __shared__ unsigned short ldsV[2][2048];   // 32 vcols x 64 keys bf16
    __shared__ unsigned short ldsP[8][1024];   // per-wave P tile 16x64

    const int tid = threadIdx.x, lane = tid & 31, wave = tid >> 5;
    const int bid = blockIdx.x;
    const int bn = bid >> 3;                       // (b,head) pair 0..31
    const int b = bn >> 3, nh = bn & 7;
    const int q0 = (bid & 7) * 128 + wave * 16;    // query tile base in [0,1024)

    const unsigned short* gK = Kbf + (size_t)bn * HW * PK;
    const unsigned short* gV = Vt + (size_t)bn * PV_ * HW;

    const v16bf aq = tileA<32>(Qbf + ((size_t)bn * QHW + q0) * PK, lane);

    v8f o0 = {0.f, 0.f, 0.f, 0.f, 0.f, 0.f, 0.f, 0.f};
    v8f o1 = o0;
    float m[8], l[8];
#pragma unroll
    for (int r = 0; r < 8; ++r) { m[r] = -1e30f; l[r] = 0.f; }

    // staging: every thread issues exactly 2 async b128 copies per chunk
    //   K chunk: 64*32 bf16 = 4KB contiguous; V chunk: 32 rows x 128B
    auto stage = [&](unsigned short* bK, unsigned short* bV, int c) {
        copy16_g2l(bK + tid * 8, gK + (size_t)c * 2048 + tid * 8);
        const int vr = tid >> 3, pc = tid & 7;
        copy16_g2l(bV + vr * 64 + pc * 8, gV + (size_t)vr * HW + c * 64 + pc * 8);
    };

    stage(ldsK[0], ldsV[0], 0);

    const int cb = lane & 15, ro = (lane & 16) ? 8 : 0;
    unsigned short* P = ldsP[wave];

    for (int c = 0; c < 64; ++c) {
        const int cur = c & 1;
        if (c + 1 < 64) { stage(ldsK[cur ^ 1], ldsV[cur ^ 1], c + 1); wait_stage<2>(); }
        else            { wait_stage<0>(); }
        __syncthreads();

        const unsigned short* Kc = ldsK[cur];
        const unsigned short* Vc = ldsV[cur];

        // ---- scores: 4 tiles of 16 keys, one WMMA each (k-dim = 32) ----
        v8f z = {0.f, 0.f, 0.f, 0.f, 0.f, 0.f, 0.f, 0.f};
        v8f s[4];
#pragma unroll
        for (int t = 0; t < 4; ++t)
            s[t] = wmma_bf16(aq, tileA<32>(Kc + t * 512, lane), z);

        // ---- online softmax: row reductions across the 16-lane group ----
        float cm[8], scl[8], rs[8];
#pragma unroll
        for (int r = 0; r < 8; ++r)
            cm[r] = fmaxf(fmaxf(s[0][r], s[1][r]), fmaxf(s[2][r], s[3][r]));
#pragma unroll
        for (int off = 1; off < 16; off <<= 1)
#pragma unroll
            for (int r = 0; r < 8; ++r) cm[r] = fmaxf(cm[r], __shfl_xor(cm[r], off, 32));
#pragma unroll
        for (int r = 0; r < 8; ++r) {
            const float nm = fmaxf(m[r], cm[r]);
            scl[r] = __expf(m[r] - nm);
            m[r] = nm;
        }
#pragma unroll
        for (int t = 0; t < 4; ++t)
#pragma unroll
            for (int r = 0; r < 8; ++r) s[t][r] = __expf(s[t][r] - m[r]);
#pragma unroll
        for (int r = 0; r < 8; ++r)
            rs[r] = (s[0][r] + s[1][r]) + (s[2][r] + s[3][r]);
#pragma unroll
        for (int off = 1; off < 16; off <<= 1)
#pragma unroll
            for (int r = 0; r < 8; ++r) rs[r] += __shfl_xor(rs[r], off, 32);
#pragma unroll
        for (int r = 0; r < 8; ++r) l[r] = l[r] * scl[r] + rs[r];

        // ---- P: C-layout -> LDS -> A-layout (per wave; LDS ops in-order) ----
#pragma unroll
        for (int t = 0; t < 4; ++t)
#pragma unroll
            for (int r = 0; r < 8; ++r)
                P[(r + ro) * 64 + cb + 16 * t] = f2bf(s[t][r]);
        asm volatile("s_wait_dscnt 0x0" ::: "memory");
        const v16bf ap0 = tileA<64>(P, lane);        // keys 0..31 of chunk
        const v16bf ap1 = tileA<64>(P + 32, lane);   // keys 32..63 of chunk

#pragma unroll
        for (int r = 0; r < 8; ++r) { o0[r] *= scl[r]; o1[r] *= scl[r]; }
        o0 = wmma_bf16(ap0, tileA<64>(Vc, lane), o0);               // vcols 0..15
        o0 = wmma_bf16(ap1, tileA<64>(Vc + 32, lane), o0);
        o1 = wmma_bf16(ap0, tileA<64>(Vc + 16 * 64, lane), o1);     // vcols 16..31
        o1 = wmma_bf16(ap1, tileA<64>(Vc + 16 * 64 + 32, lane), o1);

        __syncthreads();
    }

    // --- normalize + write out [B,32,32,N*V] f32 ---
#pragma unroll
    for (int r = 0; r < 8; ++r) {
        const float inv = 1.0f / l[r];
        const int q = q0 + r + ro;
        float* row = out + ((size_t)b * QHW + q) * NCOL + nh * PV_;
        row[cb]      = o0[r] * inv;
        row[cb + 16] = o1[r] * inv;
    }
}

// ---------------- host launcher ----------------
extern "C" void kernel_launch(void* const* d_in, const int* in_sizes, int n_in,
                              void* d_out, int out_size, void* d_ws, size_t ws_size,
                              hipStream_t stream) {
    const float* blob = (const float*)d_in[0];
    const float* wq   = (const float*)d_in[1];
    const float* bq   = (const float*)d_in[2];
    const float* wk   = (const float*)d_in[3];
    const float* bk   = (const float*)d_in[4];
    const float* wv   = (const float*)d_in[5];
    const float* bv   = (const float*)d_in[6];
    float* out = (float*)d_out;

    unsigned short* ws      = (unsigned short*)d_ws;   // ~26.4 MB used
    unsigned short* wkT     = ws + OFF_WKT;
    unsigned short* wvT     = ws + OFF_WVT;
    unsigned short* wqT     = ws + OFF_WQT;
    unsigned short* blob_bf = ws + OFF_BLOB;
    unsigned short* Kbf     = ws + OFF_K;
    unsigned short* Vt      = ws + OFF_V;
    unsigned short* Qbf     = ws + OFF_Q;

    prep_kernel<<<256, 256, 0, stream>>>(wq, wk, wv, wqT, wkT, wvT);
    blobcvt_kernel<<<4096, 256, 0, stream>>>(blob, blob_bf);
    proj_kv_kernel<<<1024, 256, 0, stream>>>(blob_bf, wkT, bk, wvT, bv, Kbf, Vt);
    qproj_kernel<<<256, 256, 0, stream>>>(blob, wqT, bq, Qbf);
    attn_kernel<<<256, 256, 0, stream>>>(Qbf, Kbf, Vt, out);
}
